// PolicyValueNet_27058293964986
// MI455X (gfx1250) — matrix operand order repaired and checked by reading, not compiled, not run on hardware
//
#include <hip/hip_runtime.h>
#include <hip/hip_bf16.h>

// ---------------------------------------------------------------------------
// PolicyValueNet on gfx1250 (MI455X): v_wmma_f32_16x16x32_f16 everywhere,
// f16 activations, LDS-staged weight panels (double buffered, bank-conflict
// padded), fused gather+concat in the policy-scorer GEMM.
// ---------------------------------------------------------------------------

#define NB   4096      // boards
#define NTM  131072    // total packed moves
#define M_CONV (NB * 64)   // 262144 rows for conv GEMMs

#define KC   64        // K-chunk staged into LDS per barrier
#define LDB  72        // padded LDS column stride in halves (36 dwords -> conflict-free)

typedef _Float16 h8   __attribute__((ext_vector_type(8)));
typedef _Float16 v16h __attribute__((ext_vector_type(16)));
typedef float    v8f  __attribute__((ext_vector_type(8)));

#define WMMA_F16(a, b, c) \
  __builtin_amdgcn_wmma_f32_16x16x32_f16(false, (a), false, (b), (short)0, (c), false, false)

// A fragment (16x32 f16): lane<16 row M=lane holds K {koff..+7, koff+16..+23},
// koff=0; lanes 16-31 same rows, koff=8.  p = row + kb + koff.
__device__ __forceinline__ v16h load_a_frag(const _Float16* __restrict__ p) {
  h8 lo = *(const h8*)(p);
  h8 hi = *(const h8*)(p + 16);
  return __builtin_shufflevector(lo, hi, 0, 1, 2, 3, 4, 5, 6, 7,
                                 8, 9, 10, 11, 12, 13, 14, 15);
}

// halves {0..7, 16..23} via two 16B loads (16B-aligned LDS friendly).
__device__ __forceinline__ v16h load_b_frag(const _Float16* __restrict__ p) {
  h8 lo = *(const h8*)(p);
  h8 hi = *(const h8*)(p + 16);
  return __builtin_shufflevector(lo, hi, 0, 1, 2, 3, 4, 5, 6, 7,
                                 8, 9, 10, 11, 12, 13, 14, 15);
}

__device__ __forceinline__ void store_col_relu(const v8f& acc, _Float16* __restrict__ C,
                                               size_t ldc, int m0, int rbase, int n,
                                               float bv) {
#pragma unroll
  for (int v = 0; v < 8; ++v) {
    float x = acc[v] + bv;
    x = x > 0.f ? x : 0.f;
    C[(size_t)(m0 + rbase + v) * ldc + n] = (_Float16)x;
  }
}

// ---------------------------------------------------------------------------
// Packing kernels (f32 -> f16, layout shuffles)
// ---------------------------------------------------------------------------

// boards [B,4,8,8] f32 NCHW -> act0 [B*64, 4] f16 NHWC
__global__ void k_pack_boards(const float* __restrict__ in, _Float16* __restrict__ out) {
  int i = blockIdx.x * blockDim.x + threadIdx.x;
  if (i >= M_CONV * 4) return;
  int c = i & 3, p = i >> 2;
  int bd = p >> 6, pix = p & 63, y = pix >> 3, x = pix & 7;
  out[i] = (_Float16)in[((bd * 4 + c) * 8 + y) * 8 + x];
}

// conv weight OIHW [O,I,3,3] f32 -> [9, O, I] f16 (per kernel position)
__global__ void k_pack_conv_w(const float* __restrict__ w, _Float16* __restrict__ out,
                              int O, int I) {
  int i = blockIdx.x * blockDim.x + threadIdx.x;
  if (i >= 9 * O * I) return;
  int pos = i / (O * I), rem = i % (O * I);
  int o = rem / I, ii = rem % I;
  int ky = pos / 3, kx = pos % 3;
  out[i] = (_Float16)w[((o * I + ii) * 3 + ky) * 3 + kx];
}

// fc_enc_w [256, 4096] f32; permute columns: reference flattens NCHW (c*64+s),
// our activations flatten NHWC (s*64+c).
__global__ void k_pack_fc_w(const float* __restrict__ w, _Float16* __restrict__ out) {
  int i = blockIdx.x * blockDim.x + threadIdx.x;
  if (i >= 256 * 4096) return;
  int n = i >> 12, colp = i & 4095;
  int s = colp >> 6, c = colp & 63;
  out[i] = (_Float16)w[n * 4096 + c * 64 + s];
}

// generic [N,K] f32 -> [N,Kpad] f16 (zero padded K)
__global__ void k_pack_w_pad(const float* __restrict__ w, _Float16* __restrict__ out,
                             int N, int K, int Kpad) {
  int i = blockIdx.x * blockDim.x + threadIdx.x;
  if (i >= N * Kpad) return;
  int n = i / Kpad, k = i % Kpad;
  out[i] = (_Float16)(k < K ? w[n * K + k] : 0.f);
}

// move_feats [TM,10] f32 -> [TM,64] f16 zero padded (K multiple of KC)
__global__ void k_pack_mf(const float* __restrict__ in, _Float16* __restrict__ out) {
  int i = blockIdx.x * blockDim.x + threadIdx.x;
  if (i >= NTM * 64) return;
  int k = i & 63, r = i >> 6;
  out[i] = (_Float16)(k < 10 ? in[r * 10 + k] : 0.f);
}

// ---------------------------------------------------------------------------
// conv1: Cin=4, too small for WMMA K=32; direct VALU. NHWC f16 in/out + ReLU.
// ---------------------------------------------------------------------------
__global__ void k_conv1(const _Float16* __restrict__ act0, const float* __restrict__ w,
                        const float* __restrict__ b, _Float16* __restrict__ out) {
  int i = blockIdx.x * blockDim.x + threadIdx.x;   // p*32 + co
  if (i >= M_CONV * 32) return;
  int co = i & 31, p = i >> 5;
  int bd = p >> 6, pix = p & 63, y = pix >> 3, x = pix & 7;
  float acc = b[co];
#pragma unroll
  for (int ky = 0; ky < 3; ++ky) {
#pragma unroll
    for (int kx = 0; kx < 3; ++kx) {
      int ys = y + ky - 1, xs = x + kx - 1;
      if ((unsigned)ys < 8u && (unsigned)xs < 8u) {
        const _Float16* src = act0 + (size_t)((bd << 6) + ys * 8 + xs) * 4;
#pragma unroll
        for (int ci = 0; ci < 4; ++ci)
          acc += (float)src[ci] * w[((co * 4 + ci) * 3 + ky) * 3 + kx];
      }
    }
  }
  out[(size_t)p * 32 + co] = (_Float16)(acc > 0.f ? acc : 0.f);
}

// ---------------------------------------------------------------------------
// Direct 3x3 conv via WMMA with LDS-staged weights.
// Block = 128 rows (8 waves x 16) x Cout 64. Per kernel position the 64xCIN
// weight panel is staged into double-buffered LDS (stage pos+1 during pos).
// ---------------------------------------------------------------------------
template <int CIN>
__global__ __launch_bounds__(256) void k_conv3x3_wmma(
    const _Float16* __restrict__ act, const _Float16* __restrict__ Wp /*[9,64,CIN]*/,
    const float* __restrict__ bias, _Float16* __restrict__ out) {
  constexpr int LDSB = CIN + 8;                 // padded col stride (halves)
  __shared__ _Float16 lds[2][64 * LDSB];

  const int tid = threadIdx.x;
  const int lane = tid & 31;
  const int waveId = tid >> 5;
  const int m0 = (blockIdx.x << 7) + (waveId << 4);

  // staging: 4 threads per column, CIN/4 halves each
  const int sc = tid >> 2;
  const int sj = (tid & 3) * (CIN / 4);

  const int p = m0 + (lane & 15);
  const int bd = p >> 6, pix = p & 63, y = pix >> 3, x = pix & 7;
  const int koff = (lane & 16) ? 8 : 0;
  const int col = lane & 15;
  const int kbh = (lane & 16) ? 8 : 0;

  // prime position 0
#pragma unroll
  for (int q = 0; q < CIN / 4; q += 8)
    *(h8*)(&lds[0][sc * LDSB + sj + q]) =
        *(const h8*)(Wp + (size_t)sc * CIN + sj + q);

  v8f acc0 = {}, acc1 = {}, acc2 = {}, acc3 = {};
  int buf = 0;
#pragma unroll
  for (int pos = 0; pos < 9; ++pos) {
    __syncthreads();
    if (pos < 8) {     // stage next position into the other buffer
      const _Float16* Ws = Wp + ((size_t)(pos + 1) * 64 + sc) * CIN + sj;
#pragma unroll
      for (int q = 0; q < CIN / 4; q += 8)
        *(h8*)(&lds[buf ^ 1][sc * LDSB + sj + q]) = *(const h8*)(Ws + q);
    }
    const int ky = pos / 3, kx = pos % 3;
    const int ys = y + ky - 1, xs = x + kx - 1;
    const bool valid = ((unsigned)ys < 8u) && ((unsigned)xs < 8u);
    const _Float16* Arow = act + (size_t)((bd << 6) + ys * 8 + xs) * CIN;
#pragma unroll
    for (int kb = 0; kb < CIN; kb += 32) {
      v16h a = {};
      if (valid) a = load_a_frag(Arow + kb + koff);
      const _Float16* Lb = &lds[buf][(size_t)col * LDSB + kbh + kb];
      v16h b0 = load_b_frag(Lb);
      v16h b1 = load_b_frag(Lb + 16 * LDSB);
      v16h b2 = load_b_frag(Lb + 32 * LDSB);
      v16h b3 = load_b_frag(Lb + 48 * LDSB);
      acc0 = WMMA_F16(a, b0, acc0);
      acc1 = WMMA_F16(a, b1, acc1);
      acc2 = WMMA_F16(a, b2, acc2);
      acc3 = WMMA_F16(a, b3, acc3);
    }
    buf ^= 1;
  }
  const int rbase = (lane & 16) ? 8 : 0;
  store_col_relu(acc0, out, 64, m0, rbase, col + 0,  bias[col + 0]);
  store_col_relu(acc1, out, 64, m0, rbase, col + 16, bias[col + 16]);
  store_col_relu(acc2, out, 64, m0, rbase, col + 32, bias[col + 32]);
  store_col_relu(acc3, out, 64, m0, rbase, col + 48, bias[col + 48]);
}

// ---------------------------------------------------------------------------
// Generic GEMM: C[M,N] = relu(A[M,K] * W[N,K]^T + bias).
// Block = 128 rows x 64 cols; weight K-chunks (KC=64) staged into
// double-buffered, bank-padded LDS shared by all 8 waves.
// Requires M%128==0, N%64==0, K%64==0 (arranged by the packing phase).
// ---------------------------------------------------------------------------
__global__ __launch_bounds__(256) void k_gemm_relu(
    const _Float16* __restrict__ A, int lda, const _Float16* __restrict__ W,
    const float* __restrict__ bias, _Float16* __restrict__ C,
    int M, int N, int K) {
  __shared__ _Float16 lds[2][64 * LDB];
  const int tid = threadIdx.x;
  const int lane = tid & 31;
  const int waveId = tid >> 5;
  const int nBlocks = N >> 6;
  const int mb = blockIdx.x / nBlocks, nb = blockIdx.x % nBlocks;
  const int n0 = nb << 6;
  const int m0 = (mb << 7) + (waveId << 4);

  const int sc = tid >> 2;        // staging column 0..63
  const int sj = (tid & 3) << 4;  // staging sub-offset 0/16/32/48
  const _Float16* Wst = W + (size_t)(n0 + sc) * K + sj;

  const int koff = (lane & 16) ? 8 : 0;
  const int col = lane & 15;
  const int kbh = (lane & 16) ? 8 : 0;
  const _Float16* Arow = A + (size_t)(m0 + (lane & 15)) * lda;

  // prime chunk 0
  *(h8*)(&lds[0][sc * LDB + sj])     = *(const h8*)(Wst);
  *(h8*)(&lds[0][sc * LDB + sj + 8]) = *(const h8*)(Wst + 8);

  v8f acc0 = {}, acc1 = {}, acc2 = {}, acc3 = {};
  int buf = 0;
  for (int kb = 0; kb < K; kb += KC) {
    __syncthreads();
    if (kb + KC < K) {
      const _Float16* Ws = Wst + kb + KC;
      *(h8*)(&lds[buf ^ 1][sc * LDB + sj])     = *(const h8*)(Ws);
      *(h8*)(&lds[buf ^ 1][sc * LDB + sj + 8]) = *(const h8*)(Ws + 8);
      __builtin_prefetch(Arow + kb + 2 * KC, 0, 1);  // global_prefetch_b8
    }
#pragma unroll
    for (int s = 0; s < KC; s += 32) {
      v16h a = load_a_frag(Arow + kb + s + koff);
      const _Float16* Lb = &lds[buf][(size_t)col * LDB + kbh + s];
      v16h b0 = load_b_frag(Lb);
      v16h b1 = load_b_frag(Lb + 16 * LDB);
      v16h b2 = load_b_frag(Lb + 32 * LDB);
      v16h b3 = load_b_frag(Lb + 48 * LDB);
      acc0 = WMMA_F16(a, b0, acc0);
      acc1 = WMMA_F16(a, b1, acc1);
      acc2 = WMMA_F16(a, b2, acc2);
      acc3 = WMMA_F16(a, b3, acc3);
    }
    buf ^= 1;
  }
  const int rbase = (lane & 16) ? 8 : 0;
  store_col_relu(acc0, C, N, m0, rbase, n0 + col + 0,  bias[n0 + col + 0]);
  store_col_relu(acc1, C, N, m0, rbase, n0 + col + 16, bias[n0 + col + 16]);
  store_col_relu(acc2, C, N, m0, rbase, n0 + col + 32, bias[n0 + col + 32]);
  store_col_relu(acc3, C, N, m0, rbase, n0 + col + 48, bias[n0 + col + 48]);
}

// ---------------------------------------------------------------------------
// Policy scorer layer 1 with FUSED gather+concat:
// joint[row] = [ emb[seg[row]] (256) , mv[row] (64) ]  -> K=320, N=128.
// Every KC chunk lies entirely in one source region (256 is KC-aligned).
// ---------------------------------------------------------------------------
__global__ __launch_bounds__(256) void k_gemm_ps1(
    const _Float16* __restrict__ emb /*[B,256]*/, const _Float16* __restrict__ mv /*[TM,64]*/,
    const int* __restrict__ seg, const _Float16* __restrict__ W /*[128,320]*/,
    const float* __restrict__ bias, _Float16* __restrict__ H /*[TM,128]*/) {
  const int K = 320, N = 128;
  __shared__ _Float16 lds[2][64 * LDB];
  const int tid = threadIdx.x;
  const int lane = tid & 31;
  const int waveId = tid >> 5;
  const int nBlocks = 2;
  const int mb = blockIdx.x / nBlocks, nb = blockIdx.x % nBlocks;
  const int n0 = nb << 6;
  const int m0 = (mb << 7) + (waveId << 4);

  const int sc = tid >> 2;
  const int sj = (tid & 3) << 4;
  const _Float16* Wst = W + (size_t)(n0 + sc) * K + sj;

  const int row = m0 + (lane & 15);
  const int s0 = seg[row];
  const _Float16* embRow = emb + (size_t)s0 * 256;
  const _Float16* mvRow = mv + (size_t)row * 64;

  const int koff = (lane & 16) ? 8 : 0;
  const int col = lane & 15;
  const int kbh = (lane & 16) ? 8 : 0;

  *(h8*)(&lds[0][sc * LDB + sj])     = *(const h8*)(Wst);
  *(h8*)(&lds[0][sc * LDB + sj + 8]) = *(const h8*)(Wst + 8);

  v8f acc0 = {}, acc1 = {}, acc2 = {}, acc3 = {};
  int buf = 0;
#pragma unroll
  for (int kb = 0; kb < K; kb += KC) {
    __syncthreads();
    if (kb + KC < K) {
      const _Float16* Ws = Wst + kb + KC;
      *(h8*)(&lds[buf ^ 1][sc * LDB + sj])     = *(const h8*)(Ws);
      *(h8*)(&lds[buf ^ 1][sc * LDB + sj + 8]) = *(const h8*)(Ws + 8);
    }
    const _Float16* src = (kb < 256) ? (embRow + kb) : (mvRow + (kb - 256));
#pragma unroll
    for (int s = 0; s < KC; s += 32) {
      v16h a = load_a_frag(src + s + koff);
      const _Float16* Lb = &lds[buf][(size_t)col * LDB + kbh + s];
      v16h b0 = load_b_frag(Lb);
      v16h b1 = load_b_frag(Lb + 16 * LDB);
      v16h b2 = load_b_frag(Lb + 32 * LDB);
      v16h b3 = load_b_frag(Lb + 48 * LDB);
      acc0 = WMMA_F16(a, b0, acc0);
      acc1 = WMMA_F16(a, b1, acc1);
      acc2 = WMMA_F16(a, b2, acc2);
      acc3 = WMMA_F16(a, b3, acc3);
    }
    buf ^= 1;
  }
  const int rbase = (lane & 16) ? 8 : 0;
  store_col_relu(acc0, H, N, m0, rbase, n0 + col + 0,  bias[n0 + col + 0]);
  store_col_relu(acc1, H, N, m0, rbase, n0 + col + 16, bias[n0 + col + 16]);
  store_col_relu(acc2, H, N, m0, rbase, n0 + col + 32, bias[n0 + col + 32]);
  store_col_relu(acc3, H, N, m0, rbase, n0 + col + 48, bias[n0 + col + 48]);
}

// ---------------------------------------------------------------------------
// N=1 head: out[r] = dot(h[r,:K], w) + b   (inputs already relu'd)
// ---------------------------------------------------------------------------
__global__ void k_head_dot(const _Float16* __restrict__ h, const float* __restrict__ w,
                           const float* __restrict__ b, float* __restrict__ out,
                           int M, int K) {
  int r = blockIdx.x * blockDim.x + threadIdx.x;
  if (r >= M) return;
  const _Float16* row = h + (size_t)r * K;
  float acc = b[0];
  for (int j = 0; j < K; j += 8) {
    h8 v = *(const h8*)(row + j);
#pragma unroll
    for (int q = 0; q < 8; ++q) acc += (float)v[q] * w[j + q];
  }
  out[r] = acc;
}

// ---------------------------------------------------------------------------
// Host side
// ---------------------------------------------------------------------------
static inline int gblk(long long threads) { return (int)((threads + 255) / 256); }

extern "C" void kernel_launch(void* const* d_in, const int* in_sizes, int n_in,
                              void* d_out, int out_size, void* d_ws, size_t ws_size,
                              hipStream_t stream) {
  const float* boards  = (const float*)d_in[0];
  const float* mfeats  = (const float*)d_in[1];
  const int*   seg     = (const int*)d_in[2];
  const float* c1w = (const float*)d_in[3],  *c1b = (const float*)d_in[4];
  const float* c2w = (const float*)d_in[5],  *c2b = (const float*)d_in[6];
  const float* c3w = (const float*)d_in[7],  *c3b = (const float*)d_in[8];
  const float* fcw = (const float*)d_in[9],  *fcb = (const float*)d_in[10];
  const float* v1w = (const float*)d_in[11], *v1b = (const float*)d_in[12];
  const float* v2w = (const float*)d_in[13], *v2b = (const float*)d_in[14];
  const float* m1w = (const float*)d_in[15], *m1b = (const float*)d_in[16];
  const float* m2w = (const float*)d_in[17], *m2b = (const float*)d_in[18];
  const float* p1w = (const float*)d_in[19], *p1b = (const float*)d_in[20];
  const float* p2w = (const float*)d_in[21], *p2b = (const float*)d_in[22];
  float* out = (float*)d_out;          // [TM logits | B values]

  // workspace carve-out (f16 buffers, 256B aligned)
  size_t off = 0;
  auto alloc = [&](size_t halves) -> _Float16* {
    _Float16* p = (_Float16*)((char*)d_ws + off);
    off += ((halves * sizeof(_Float16) + 255) & ~(size_t)255);
    return p;
  };
  _Float16* act0   = alloc((size_t)M_CONV * 4);    // packed boards NHWC
  _Float16* act1   = alloc((size_t)M_CONV * 32);   // conv1 out
  _Float16* act2   = alloc((size_t)M_CONV * 64);   // conv2 out
  _Float16* act3   = alloc((size_t)M_CONV * 64);   // conv3 out == fc input [B,4096]
  _Float16* emb    = alloc((size_t)NB * 256);
  _Float16* vh1    = alloc((size_t)NB * 128);
  _Float16* mf     = alloc((size_t)NTM * 64);      // padded move feats (K=64)
  _Float16* mm1    = alloc((size_t)NTM * 64);
  _Float16* mm2    = alloc((size_t)NTM * 64);
  _Float16* hbuf   = alloc((size_t)NTM * 128);
  _Float16* wp_c2  = alloc(9 * 64 * 32);
  _Float16* wp_c3  = alloc(9 * 64 * 64);
  _Float16* wp_fc  = alloc((size_t)256 * 4096);
  _Float16* wp_v1  = alloc(128 * 256);
  _Float16* wp_m1  = alloc(64 * 64);
  _Float16* wp_m2  = alloc(64 * 64);
  _Float16* wp_p1  = alloc(128 * 320);
  (void)ws_size; (void)n_in; (void)in_sizes; (void)out_size;

  // ---- pack phase ----
  k_pack_boards<<<gblk((long long)M_CONV * 4), 256, 0, stream>>>(boards, act0);
  k_pack_conv_w<<<gblk(9 * 64 * 32), 256, 0, stream>>>(c2w, wp_c2, 64, 32);
  k_pack_conv_w<<<gblk(9 * 64 * 64), 256, 0, stream>>>(c3w, wp_c3, 64, 64);
  k_pack_fc_w<<<gblk((long long)256 * 4096), 256, 0, stream>>>(fcw, wp_fc);
  k_pack_w_pad<<<gblk(128 * 256), 256, 0, stream>>>(v1w, wp_v1, 128, 256, 256);
  k_pack_w_pad<<<gblk(64 * 64), 256, 0, stream>>>(m1w, wp_m1, 64, 10, 64);
  k_pack_w_pad<<<gblk(64 * 64), 256, 0, stream>>>(m2w, wp_m2, 64, 64, 64);
  k_pack_w_pad<<<gblk(128 * 320), 256, 0, stream>>>(p1w, wp_p1, 128, 320, 320);
  k_pack_mf<<<gblk((long long)NTM * 64), 256, 0, stream>>>(mfeats, mf);

  // ---- board encoder ----
  k_conv1<<<gblk((long long)M_CONV * 32), 256, 0, stream>>>(act0, c1w, c1b, act1);
  k_conv3x3_wmma<32><<<M_CONV / 128, 256, 0, stream>>>(act1, wp_c2, c2b, act2);
  k_conv3x3_wmma<64><<<M_CONV / 128, 256, 0, stream>>>(act2, wp_c3, c3b, act3);
  // fc_enc: A = act3 as [B, 4096] (NHWC flatten; wp_fc columns permuted to match)
  k_gemm_relu<<<(NB / 128) * (256 / 64), 256, 0, stream>>>(
      act3, 4096, wp_fc, fcb, emb, NB, 256, 4096);

  // ---- value head (relu(emb)==emb since emb already relu'd) ----
  k_gemm_relu<<<(NB / 128) * (128 / 64), 256, 0, stream>>>(
      emb, 256, wp_v1, v1b, vh1, NB, 128, 256);
  k_head_dot<<<gblk(NB), 256, 0, stream>>>(vh1, v2w, v2b, out + NTM, NB, 128);

  // ---- move MLP ----
  k_gemm_relu<<<(NTM / 128) * 1, 256, 0, stream>>>(mf, 64, wp_m1, m1b, mm1, NTM, 64, 64);
  k_gemm_relu<<<(NTM / 128) * 1, 256, 0, stream>>>(mm1, 64, wp_m2, m2b, mm2, NTM, 64, 64);

  // ---- joint policy scorer (fused gather+concat) ----
  k_gemm_ps1<<<(NTM / 128) * 2, 256, 0, stream>>>(emb, mm2, seg, wp_p1, p1b, hbuf);
  k_head_dot<<<gblk(NTM), 256, 0, stream>>>(hbuf, p2w, p2b, out, NTM, 128);
}
